// LaneGNN_52664888983603
// MI455X (gfx1250) — matrix operand order repaired and checked by reading, not compiled, not run on hardware
//
#include <hip/hip_runtime.h>
#include <hip/hip_bf16.h>

typedef __attribute__((ext_vector_type(16))) __bf16 v16bf;
typedef __attribute__((ext_vector_type(8)))  __bf16 v8bf;
typedef __attribute__((ext_vector_type(4)))  __bf16 v4bf;
typedef __attribute__((ext_vector_type(8)))  float  v8f;

// ---------------------------------------------------------------------------
// Fragment loaders (CDNA5 ISA 7.12.2 wave32 layouts).
// A staged row-major [16 x ldA] bf16 in LDS: per-lane data is 8+8 contiguous
// halves -> two 16B LDS loads.
// W staged FRAGMENT-MAJOR in LDS: for fragment (kt,nt), lane's 16 values are
// contiguous -> one 32B LDS load, no packing movs.
// ---------------------------------------------------------------------------
__device__ __forceinline__ v16bf load_a_frag(const __bf16* sA, int ldA, int k0) {
  const int lane = threadIdx.x & 31;
  const int l = lane & 15, hf = lane >> 4;
  const __bf16* p = sA + l * ldA + k0 + hf * 8;
  const v8bf lo = *(const v8bf*)p;          // K = k0 + hf*8 + {0..7}
  const v8bf hi = *(const v8bf*)(p + 16);   // K = k0 + hf*8 + {16..23}
  return __builtin_shufflevector(lo, hi, 0, 1, 2, 3, 4, 5, 6, 7,
                                 8, 9, 10, 11, 12, 13, 14, 15);
}

__device__ __forceinline__ v16bf load_b_frag(const __bf16* sWf, int ntiles,
                                             int kt, int nt) {
  const int lane = threadIdx.x & 31;
  return *(const v16bf*)(sWf + (((kt * ntiles + nt) << 5) + lane) * 16);
}

__device__ __forceinline__ v8f wmma_tile(const __bf16* sA, int ldA,
                                         const __bf16* sWf, int ntiles, int nt,
                                         int K, v8f acc) {
  for (int k0 = 0; k0 < K; k0 += 32) {
    const v16bf a = load_a_frag(sA, ldA, k0);
    const v16bf b = load_b_frag(sWf, ntiles, k0 >> 5, nt);
    acc = __builtin_amdgcn_wmma_f32_16x16x32_bf16(false, a, false, b,
                                                  (short)0, acc, false, false);
  }
  return acc;
}

// One hidden layer for a 16-row tile: out = relu(A @ W + b) -> LDS bf16.
// Safe for sOut == sA (all WMMA reads precede stores in wave program order;
// DS ops are in-order within a wave).
template <int K, int NOUT>
__device__ __forceinline__ void layer_relu_lds(const __bf16* sA, const __bf16* sWf,
                                               const float* sB, __bf16* sOut) {
  const int lane = threadIdx.x & 31;
  const int l = lane & 15, hf = lane >> 4;
  constexpr int NT = NOUT / 16;
  v8f acc[NT];
#pragma unroll
  for (int t = 0; t < NT; ++t) {
    v8f z = {0.f, 0.f, 0.f, 0.f, 0.f, 0.f, 0.f, 0.f};
    acc[t] = wmma_tile(sA, K, sWf, NT, t, K, z);
  }
  __builtin_amdgcn_wave_barrier();
#pragma unroll
  for (int t = 0; t < NT; ++t) {
    const float bias = sB[t * 16 + l];
#pragma unroll
    for (int r = 0; r < 8; ++r)
      sOut[(r + hf * 8) * NOUT + t * 16 + l] = (__bf16)fmaxf(acc[t][r] + bias, 0.f);
  }
  __builtin_amdgcn_wave_barrier();
}

// Stage W[K x N] (row-major f32) into fragment-major bf16 LDS; rows >= Keff
// are zero-filled (K padding).
__device__ __forceinline__ void stage_w_frag(const float* g, __bf16* s,
                                             int Kpad, int Keff, int N) {
  const int total = Kpad * N;
  const int ntiles = N >> 4;
  for (int j = threadIdx.x; j < total; j += blockDim.x) {
    const int i = j & 15;
    const int lane = (j >> 4) & 31;
    const int frag = j >> 9;
    const int nt = frag % ntiles;
    const int kt = frag / ntiles;
    const int k = (kt << 5) + ((lane >> 4) << 3) + (i & 7) + ((i >> 3) << 4);
    const int n = (nt << 4) + (lane & 15);
    s[j] = (k < Keff) ? (__bf16)g[k * N + n] : (__bf16)0.0f;
  }
}

__device__ __forceinline__ void stage_b(const float* g, float* s, int n) {
  for (int i = threadIdx.x; i < n; i += blockDim.x) s[i] = g[i];
}

__device__ __forceinline__ void store4(__bf16* dst, float4 q) {
  v4bf v = {(__bf16)q.x, (__bf16)q.y, (__bf16)q.z, (__bf16)q.w};
  *(v4bf*)dst = v;
}

// ---------------------------------------------------------------------------
__global__ void zero_kernel(float* __restrict__ p, int n) {
  const int i = blockIdx.x * blockDim.x + threadIdx.x;
  if (i < n) p[i] = 0.f;
}

// pose_encoder: [2 -> 16 -> 32], VALU (K too small for WMMA).
__global__ void pose_kernel(const float* __restrict__ feats,
                            const float* __restrict__ W0, const float* __restrict__ B0,
                            const float* __restrict__ W1, const float* __restrict__ B1,
                            float* __restrict__ x, int N) {
  const int n = blockIdx.x * blockDim.x + threadIdx.x;
  if (n >= N) return;
  const float f0 = feats[n * 2], f1 = feats[n * 2 + 1];
  float h[16];
#pragma unroll
  for (int j = 0; j < 16; ++j)
    h[j] = fmaxf(f0 * W0[j] + f1 * W0[16 + j] + B0[j], 0.f);
#pragma unroll
  for (int o = 0; o < 32; ++o) {
    float a = B1[o];
#pragma unroll
    for (int j = 0; j < 16; ++j) a += h[j] * W1[j * 32 + o];
    x[n * 32 + o] = a;
  }
}

// classifier heads: [32 -> 16 -> 8 -> 1], VALU.
__global__ void head_kernel(const float* __restrict__ in,
                            const float* __restrict__ W0, const float* __restrict__ B0,
                            const float* __restrict__ W1, const float* __restrict__ B1,
                            const float* __restrict__ W2, const float* __restrict__ B2,
                            float* __restrict__ out, int R) {
  const int r = blockIdx.x * blockDim.x + threadIdx.x;
  if (r >= R) return;
  float v[32];
#pragma unroll
  for (int i = 0; i < 8; ++i) {
    const float4 q = ((const float4*)(in + (size_t)r * 32))[i];
    v[i * 4] = q.x; v[i * 4 + 1] = q.y; v[i * 4 + 2] = q.z; v[i * 4 + 3] = q.w;
  }
  float h1[16];
#pragma unroll
  for (int j = 0; j < 16; ++j) {
    float a = B0[j];
#pragma unroll
    for (int k = 0; k < 32; ++k) a += v[k] * W0[k * 16 + j];
    h1[j] = fmaxf(a, 0.f);
  }
  float h2[8];
#pragma unroll
  for (int j = 0; j < 8; ++j) {
    float a = B1[j];
#pragma unroll
    for (int k = 0; k < 16; ++k) a += h1[k] * W1[k * 8 + j];
    h2[j] = fmaxf(a, 0.f);
  }
  float a = B2[0];
#pragma unroll
  for (int k = 0; k < 8; ++k) a += h2[k] * W2[k];
  out[r] = a;
}

// edge_encoder [4->8->16] (VALU) fused with fuse_edge [80->64->32] (WMMA,
// K padded 80->96 with zeros on both A and W). 4 waves/block, 16 edges/wave.
__global__ void __launch_bounds__(128) fuse_edge_kernel(
    const float* __restrict__ geo, const float* __restrict__ mapf,
    const float* __restrict__ eW0, const float* __restrict__ eB0,
    const float* __restrict__ eW1, const float* __restrict__ eB1,
    const float* __restrict__ fW0, const float* __restrict__ fB0,
    const float* __restrict__ fW1, const float* __restrict__ fB1,
    float* __restrict__ ea, int E) {
  __shared__ alignas(32) __bf16 sW1[96 * 64];
  __shared__ alignas(32) __bf16 sW2[64 * 32];
  __shared__ float  sB1[64], sB2[32];
  __shared__ alignas(32) __bf16 sA[4][16 * 96];
  __shared__ alignas(32) __bf16 sH[4][16 * 64];

  stage_w_frag(fW0, sW1, 96, 80, 64);
  stage_w_frag(fW1, sW2, 64, 64, 32);
  stage_b(fB0, sB1, 64);
  stage_b(fB1, sB2, 32);
  __syncthreads();

  const int wave = threadIdx.x >> 5, lane = threadIdx.x & 31;
  const int l = lane & 15, hf = lane >> 4;
  const int ebase = blockIdx.x * 64 + wave * 16;
  if (ebase >= E) return;
  __bf16* A = sA[wave];
  __bf16* H = sH[wave];

  if (hf == 0) {  // one lane per edge: tiny edge_encoder MLP on VALU
    const int e = ebase + l;
    const float g0 = geo[e * 4 + 0], g1 = geo[e * 4 + 1];
    const float g2 = geo[e * 4 + 2], g3 = geo[e * 4 + 3];
    float h8[8];
#pragma unroll
    for (int j = 0; j < 8; ++j)
      h8[j] = fmaxf(eB0[j] + g0 * eW0[j] + g1 * eW0[8 + j] +
                    g2 * eW0[16 + j] + g3 * eW0[24 + j], 0.f);
    float o[16];
#pragma unroll
    for (int c = 0; c < 16; ++c) {
      float a = eB1[c];
#pragma unroll
      for (int j = 0; j < 8; ++j) a += h8[j] * eW1[j * 16 + c];
      o[c] = a;
    }
#pragma unroll
    for (int q = 0; q < 4; ++q) {
      v4bf v = {(__bf16)o[q * 4], (__bf16)o[q * 4 + 1],
                (__bf16)o[q * 4 + 2], (__bf16)o[q * 4 + 3]};
      *(v4bf*)(A + l * 96 + q * 4) = v;
    }
  }
  // map features -> cols 16..79 (16 float4 chunks per edge)
  for (int t = lane; t < 16 * 16; t += 32) {
    const int e = t >> 4, c4 = (t & 15) << 2;
    store4(A + e * 96 + 16 + c4,
           *(const float4*)(mapf + (ebase + e) * 64 + c4));
  }
  // zero K-pad cols 80..95 (4 chunks per edge)
  for (int t = lane; t < 16 * 4; t += 32) {
    const float4 z = {0.f, 0.f, 0.f, 0.f};
    store4(A + (t >> 2) * 96 + 80 + ((t & 3) << 2), z);
  }
  __builtin_amdgcn_wave_barrier();

  layer_relu_lds<96, 64>(A, sW1, sB1, H);

  const v8f z = {0.f, 0.f, 0.f, 0.f, 0.f, 0.f, 0.f, 0.f};
  const v8f o0 = wmma_tile(H, 64, sW2, 2, 0, 64, z);
  const v8f o1 = wmma_tile(H, 64, sW2, 2, 1, 64, z);
#pragma unroll
  for (int r = 0; r < 8; ++r) {
    const int e = ebase + r + hf * 8;
    ea[e * 32 + l]      = o0[r] + sB2[l];
    ea[e * 32 + 16 + l] = o1[r] + sB2[16 + l];
  }
}

// edge_update: e_new = MLP[96->64->64->32](x_i|x_j|ea), ea updated in place.
__global__ void __launch_bounds__(128) edge_update_kernel(
    const float* __restrict__ x,
    const int* __restrict__ rows, const int* __restrict__ cols,
    const float* __restrict__ W0, const float* __restrict__ B0,
    const float* __restrict__ W1, const float* __restrict__ B1,
    const float* __restrict__ W2, const float* __restrict__ B2,
    float* __restrict__ ea, int E) {
  __shared__ alignas(32) __bf16 sW0[96 * 64], sW1[64 * 64], sW2[64 * 32];
  __shared__ float  sB0[64], sB1[64], sB2[32];
  __shared__ alignas(32) __bf16 sA[4][16 * 96];
  __shared__ alignas(32) __bf16 sH[4][16 * 64];
  __shared__ int    sIdx[4][32];

  stage_w_frag(W0, sW0, 96, 96, 64);
  stage_w_frag(W1, sW1, 64, 64, 64);
  stage_w_frag(W2, sW2, 64, 64, 32);
  stage_b(B0, sB0, 64); stage_b(B1, sB1, 64); stage_b(B2, sB2, 32);
  __syncthreads();

  const int wave = threadIdx.x >> 5, lane = threadIdx.x & 31;
  const int l = lane & 15, hf = lane >> 4;
  const int ebase = blockIdx.x * 64 + wave * 16;
  if (ebase >= E) return;
  __bf16* A = sA[wave]; __bf16* H = sH[wave]; int* idx = sIdx[wave];

  if (hf == 0) { idx[l] = cols[ebase + l]; idx[16 + l] = rows[ebase + l]; }
  __builtin_amdgcn_wave_barrier();

  // A = [x_i | x_j | ea] as 24 float4 chunks per edge
  for (int t = lane; t < 16 * 24; t += 32) {
    const int e = t / 24, c4 = (t % 24) << 2;
    float4 q;
    if (c4 < 32)      q = *(const float4*)(x + idx[e] * 32 + c4);
    else if (c4 < 64) q = *(const float4*)(x + idx[16 + e] * 32 + (c4 - 32));
    else              q = *(const float4*)(ea + (ebase + e) * 32 + (c4 - 64));
    store4(A + e * 96 + c4, q);
  }
  __builtin_amdgcn_wave_barrier();

  layer_relu_lds<96, 64>(A, sW0, sB0, H);
  layer_relu_lds<64, 64>(H, sW1, sB1, H);   // in place

  const v8f z = {0.f, 0.f, 0.f, 0.f, 0.f, 0.f, 0.f, 0.f};
  const v8f o0 = wmma_tile(H, 64, sW2, 2, 0, 64, z);
  const v8f o1 = wmma_tile(H, 64, sW2, 2, 1, 64, z);
#pragma unroll
  for (int r = 0; r < 8; ++r) {
    const int e = ebase + r + hf * 8;
    ea[e * 32 + l]      = o0[r] + sB2[l];
    ea[e * 32 + 16 + l] = o1[r] + sB2[16 + l];
  }
}

// future/past messages [96->64->32] + fp32 atomic segment-sum scatter.
__global__ void __launch_bounds__(128) msgs_kernel(
    const float* __restrict__ x, const float* __restrict__ x0,
    const float* __restrict__ ea,
    const int* __restrict__ rows, const int* __restrict__ cols,
    const float* __restrict__ fW0, const float* __restrict__ fB0,
    const float* __restrict__ fW1, const float* __restrict__ fB1,
    const float* __restrict__ pW0, const float* __restrict__ pB0,
    const float* __restrict__ pW1, const float* __restrict__ pB1,
    float* __restrict__ msgP, float* __restrict__ msgF, int E) {
  __shared__ alignas(32) __bf16 sfW0[96 * 64], sfW1[64 * 32];
  __shared__ alignas(32) __bf16 spW0[96 * 64], spW1[64 * 32];
  __shared__ float  sfB0[64], sfB1[32], spB0[64], spB1[32];
  __shared__ alignas(32) __bf16 sA[4][16 * 96];
  __shared__ alignas(32) __bf16 sH[4][16 * 64];
  __shared__ int    sIdx[4][32];

  stage_w_frag(fW0, sfW0, 96, 96, 64);
  stage_w_frag(fW1, sfW1, 64, 64, 32);
  stage_w_frag(pW0, spW0, 96, 96, 64);
  stage_w_frag(pW1, spW1, 64, 64, 32);
  stage_b(fB0, sfB0, 64); stage_b(fB1, sfB1, 32);
  stage_b(pB0, spB0, 64); stage_b(pB1, spB1, 32);
  __syncthreads();

  const int wave = threadIdx.x >> 5, lane = threadIdx.x & 31;
  const int l = lane & 15, hf = lane >> 4;
  const int ebase = blockIdx.x * 64 + wave * 16;
  if (ebase >= E) return;
  __bf16* A = sA[wave]; __bf16* H = sH[wave]; int* idx = sIdx[wave];

  if (hf == 0) { idx[l] = cols[ebase + l]; idx[16 + l] = rows[ebase + l]; }
  __builtin_amdgcn_wave_barrier();

  // future: [x_i | e_new | x0_i] -> scatter-add msgF[rows]
  for (int t = lane; t < 16 * 24; t += 32) {
    const int e = t / 24, c4 = (t % 24) << 2;
    float4 q;
    if (c4 < 32)      q = *(const float4*)(x + idx[e] * 32 + c4);
    else if (c4 < 64) q = *(const float4*)(ea + (ebase + e) * 32 + (c4 - 32));
    else              q = *(const float4*)(x0 + idx[e] * 32 + (c4 - 64));
    store4(A + e * 96 + c4, q);
  }
  __builtin_amdgcn_wave_barrier();
  layer_relu_lds<96, 64>(A, sfW0, sfB0, H);
  {
    const v8f z = {0.f, 0.f, 0.f, 0.f, 0.f, 0.f, 0.f, 0.f};
    const v8f o0 = wmma_tile(H, 64, sfW1, 2, 0, 64, z);
    const v8f o1 = wmma_tile(H, 64, sfW1, 2, 1, 64, z);
#pragma unroll
    for (int r = 0; r < 8; ++r) {
      const int dst = idx[16 + r + hf * 8];       // rows
      unsafeAtomicAdd(&msgF[dst * 32 + l],      o0[r] + sfB1[l]);
      unsafeAtomicAdd(&msgF[dst * 32 + 16 + l], o1[r] + sfB1[16 + l]);
    }
  }
  __builtin_amdgcn_wave_barrier();

  // past: [x_j | e_new | x0_j] -> scatter-add msgP[cols]
  for (int t = lane; t < 16 * 24; t += 32) {
    const int e = t / 24, c4 = (t % 24) << 2;
    float4 q;
    if (c4 < 32)      q = *(const float4*)(x + idx[16 + e] * 32 + c4);
    else if (c4 < 64) q = *(const float4*)(ea + (ebase + e) * 32 + (c4 - 32));
    else              q = *(const float4*)(x0 + idx[16 + e] * 32 + (c4 - 64));
    store4(A + e * 96 + c4, q);
  }
  __builtin_amdgcn_wave_barrier();
  layer_relu_lds<96, 64>(A, spW0, spB0, H);
  {
    const v8f z = {0.f, 0.f, 0.f, 0.f, 0.f, 0.f, 0.f, 0.f};
    const v8f o0 = wmma_tile(H, 64, spW1, 2, 0, 64, z);
    const v8f o1 = wmma_tile(H, 64, spW1, 2, 1, 64, z);
#pragma unroll
    for (int r = 0; r < 8; ++r) {
      const int dst = idx[r + hf * 8];            // cols
      unsafeAtomicAdd(&msgP[dst * 32 + l],      o0[r] + spB1[l]);
      unsafeAtomicAdd(&msgP[dst * 32 + 16 + l], o1[r] + spB1[16 + l]);
    }
  }
}

// combine_future_past: x_new = MLP[64->64->32](msg_past|msg_future)
__global__ void __launch_bounds__(128) combine_kernel(
    const float* __restrict__ msgP, const float* __restrict__ msgF,
    const float* __restrict__ W0, const float* __restrict__ B0,
    const float* __restrict__ W1, const float* __restrict__ B1,
    float* __restrict__ xout, int N) {
  __shared__ alignas(32) __bf16 sW0[64 * 64], sW1[64 * 32];
  __shared__ float  sB0[64], sB1[32];
  __shared__ alignas(32) __bf16 sA[4][16 * 64];
  __shared__ alignas(32) __bf16 sH[4][16 * 64];
  stage_w_frag(W0, sW0, 64, 64, 64);
  stage_w_frag(W1, sW1, 64, 64, 32);
  stage_b(B0, sB0, 64); stage_b(B1, sB1, 32);
  __syncthreads();

  const int wave = threadIdx.x >> 5, lane = threadIdx.x & 31;
  const int l = lane & 15, hf = lane >> 4;
  const int nbase = blockIdx.x * 64 + wave * 16;
  if (nbase >= N) return;
  __bf16* A = sA[wave]; __bf16* H = sH[wave];

  for (int t = lane; t < 16 * 16; t += 32) {   // 16 float4 chunks per node
    const int e = t >> 4, c4 = (t & 15) << 2;
    const float4 q = (c4 < 32)
        ? *(const float4*)(msgP + (nbase + e) * 32 + c4)
        : *(const float4*)(msgF + (nbase + e) * 32 + (c4 - 32));
    store4(A + e * 64 + c4, q);
  }
  __builtin_amdgcn_wave_barrier();

  layer_relu_lds<64, 64>(A, sW0, sB0, H);
  const v8f z = {0.f, 0.f, 0.f, 0.f, 0.f, 0.f, 0.f, 0.f};
  const v8f o0 = wmma_tile(H, 64, sW1, 2, 0, 64, z);
  const v8f o1 = wmma_tile(H, 64, sW1, 2, 1, 64, z);
#pragma unroll
  for (int r = 0; r < 8; ++r) {
    const int n = nbase + r + hf * 8;
    xout[n * 32 + l]      = o0[r] + sB1[l];
    xout[n * 32 + 16 + l] = o1[r] + sB1[16 + l];
  }
}

// ---------------------------------------------------------------------------
extern "C" void kernel_launch(void* const* d_in, const int* in_sizes, int n_in,
                              void* d_out, int out_size, void* d_ws, size_t ws_size,
                              hipStream_t stream) {
  (void)n_in; (void)out_size; (void)ws_size;
  const int N = in_sizes[0] / 2;
  const int E = in_sizes[1] / 4;
  const float* feats = (const float*)d_in[0];
  const float* geo   = (const float*)d_in[1];
  const float* mapf  = (const float*)d_in[2];
  const int*   eidx  = (const int*)d_in[3];
  const int*   rows  = eidx;        // edge_index[0]
  const int*   cols  = eidx + E;    // edge_index[1]
#define P(i) ((const float*)d_in[i])
  // param leaf order (insertion order of params dict, (W,b) per layer):
  // 4..7 pose | 8..11 edge_enc | 12..15 fuse | 16..21 edge_update
  // 22..25 create_past | 26..29 create_future | 30..33 combine
  // 34..39 node_cls | 40..45 endpoint_cls | 46..51 edge_cls

  float* ws  = (float*)d_ws;
  float* x0  = ws;                          // [N,32] initial_x
  float* xA  = x0 + (size_t)N * 32;         // [N,32]
  float* xB  = xA + (size_t)N * 32;         // [N,32]
  float* ea  = xB + (size_t)N * 32;         // [E,32]
  float* msgP = ea + (size_t)E * 32;        // [N,32]
  float* msgF = msgP + (size_t)N * 32;      // [N,32]
  float* out = (float*)d_out;               // [E + N + N]

  pose_kernel<<<(N + 255) / 256, 256, 0, stream>>>(
      feats, P(4), P(5), P(6), P(7), x0, N);

  fuse_edge_kernel<<<(E + 63) / 64, 128, 0, stream>>>(
      geo, mapf, P(8), P(9), P(10), P(11), P(12), P(13), P(14), P(15), ea, E);

  const float* xin = x0;
  float* xouts[2] = {xA, xB};
  for (int s = 0; s < 2; ++s) {
    zero_kernel<<<(2 * N * 32 + 255) / 256, 256, 0, stream>>>(msgP, 2 * N * 32);
    edge_update_kernel<<<(E + 63) / 64, 128, 0, stream>>>(
        xin, rows, cols, P(16), P(17), P(18), P(19), P(20), P(21), ea, E);
    msgs_kernel<<<(E + 63) / 64, 128, 0, stream>>>(
        xin, x0, ea, rows, cols,
        P(26), P(27), P(28), P(29),   // create_future_msgs
        P(22), P(23), P(24), P(25),   // create_past_msgs
        msgP, msgF, E);
    combine_kernel<<<(N + 63) / 64, 128, 0, stream>>>(
        msgP, msgF, P(30), P(31), P(32), P(33), xouts[s], N);
    xin = xouts[s];
  }

  head_kernel<<<(E + 255) / 256, 256, 0, stream>>>(
      ea, P(46), P(47), P(48), P(49), P(50), P(51), out, E);            // edge
  head_kernel<<<(N + 255) / 256, 256, 0, stream>>>(
      xB, P(34), P(35), P(36), P(37), P(38), P(39), out + E, N);        // node
  head_kernel<<<(N + 255) / 256, 256, 0, stream>>>(
      xB, P(40), P(41), P(42), P(43), P(44), P(45), out + E + N, N);    // endpoint
#undef P
}
